// DotProductAttention_30829275250888
// MI455X (gfx1250) — matrix operand order repaired and checked by reading, not compiled, not run on hardware
//
#include <hip/hip_runtime.h>

typedef __attribute__((ext_vector_type(16))) _Float16 v16h;
typedef __attribute__((ext_vector_type(8)))  _Float16 v8h;
typedef __attribute__((ext_vector_type(2)))  _Float16 h2;
typedef __attribute__((ext_vector_type(8)))  float    v8f;

#define BB   32
#define LL   2048
#define DD   64
#define KT   64          // keys per tile
#define VSTR 72          // padded halves per sVt row (144B, 16B-aligned)

// log2(e)/sqrt(64): scores computed directly in exp2 domain
#define QSCALE (0.125f * 1.44269504088896340736f)

// DPP16 row ops never cross a 16-lane row -> exactly our reduction scope (wave32: 2 rows)
#define DPPF(x, ctrl) \
    __int_as_float(__builtin_amdgcn_update_dpp(0, __float_as_int(x), (ctrl), 0xF, 0xF, true))

__device__ __forceinline__ float red_max16(float v) {
    v = fmaxf(v, DPPF(v, 0xB1));   // quad_perm(1,0,3,2)  : xor 1
    v = fmaxf(v, DPPF(v, 0x4E));   // quad_perm(2,3,0,1)  : xor 2
    v = fmaxf(v, DPPF(v, 0x141));  // row_half_mirror     : cross 4-groups
    v = fmaxf(v, DPPF(v, 0x140));  // row_mirror          : cross 8-groups
    return v;
}

__global__ __launch_bounds__(128) void fa_fwd_kernel(
    const float* __restrict__ Q, const float* __restrict__ K,
    const float* __restrict__ V, const int* __restrict__ vlens,
    float* __restrict__ Out)
{
    __shared__ _Float16 sK[KT * DD];          // [key][d] row-major f16
    __shared__ _Float16 sVt[DD * VSTR];       // [d][key-slot] transposed f16 (sigma order)
    __shared__ _Float16 sP[4][16 * KT];       // per-wave P transpose scratch [row][slot]

    const int tid  = threadIdx.x;
    const int lane = tid & 31;
    const int wave = tid >> 5;
    const int m    = lane & 15;   // A/C row or B/C col index
    const int hi   = lane >> 4;   // half-wave select

    const int b     = blockIdx.x >> 5;        // 32 q-blocks per batch
    const int qblk  = blockIdx.x & 31;
    const int qbase = qblk * 64 + wave * 16;

    const int vlen   = vlens[b];              // 1..L
    const int ntiles = (vlen + KT - 1) >> 6;  // skip fully-masked tiles (exact: exp underflows)

    // staging assignment: rows (j, j+16) of a 32-row half so sigma-slots (2a, 2a+1) pack
    const int g  = tid >> 6;                  // which 32-key half of the tile
    const int a  = (tid >> 2) & 15;
    const int cc = (tid & 3) * 16;            // 16-wide d chunk
    const int r0 = 32 * g + a;
    const int r1 = 32 * g + a + 16;
    const int sv = 32 * g + 2 * a;            // packed key-slot for V

    // ---- Q fragments (A layout), scale folded in ----
    const float* qrow = Q + ((size_t)b * LL + qbase + m) * DD;
    v16h aq0, aq1;
    #pragma unroll
    for (int h = 0; h < 16; ++h) {
        const int k = 16 * hi + h;
        aq0[h] = (_Float16)(qrow[k]      * QSCALE);
        aq1[h] = (_Float16)(qrow[32 + k] * QSCALE);
    }

    // all-ones B tile: o4 = P * 1 accumulates the softmax denominator on the matrix pipe
    v16h ones;
    #pragma unroll
    for (int h = 0; h < 16; ++h) ones[h] = (_Float16)1.0f;

    float m_run[8];
    v8f o0 = {}, o1 = {}, o2 = {}, o3 = {}, o4 = {};
    #pragma unroll
    for (int r = 0; r < 8; ++r) m_run[r] = -3.0e38f;

    for (int t = 0; t < ntiles; ++t) {
        const int kb = t * KT;
        __syncthreads();   // previous tile's LDS fully consumed
        {   // ---- stage K (row-major) and V (transposed, sigma slot order), f32->f16 ----
            const size_t base = (size_t)b * LL + kb;
            const float* kp0 = K + (base + r0) * DD + cc;
            const float* kp1 = K + (base + r1) * DD + cc;
            const float* vp0 = V + (base + r0) * DD + cc;
            const float* vp1 = V + (base + r1) * DD + cc;
            float k0[16], k1[16], w0[16], w1[16];
            #pragma unroll
            for (int i = 0; i < 4; ++i) {
                *(float4*)&k0[4*i] = ((const float4*)kp0)[i];
                *(float4*)&k1[4*i] = ((const float4*)kp1)[i];
                *(float4*)&w0[4*i] = ((const float4*)vp0)[i];
                *(float4*)&w1[4*i] = ((const float4*)vp1)[i];
            }
            v8h pa, pb_;
            #pragma unroll
            for (int j = 0; j < 8; ++j) { pa[j] = (_Float16)k0[j]; pb_[j] = (_Float16)k0[8+j]; }
            *(v8h*)&sK[r0 * DD + cc]     = pa;
            *(v8h*)&sK[r0 * DD + cc + 8] = pb_;
            #pragma unroll
            for (int j = 0; j < 8; ++j) { pa[j] = (_Float16)k1[j]; pb_[j] = (_Float16)k1[8+j]; }
            *(v8h*)&sK[r1 * DD + cc]     = pa;
            *(v8h*)&sK[r1 * DD + cc + 8] = pb_;
            #pragma unroll
            for (int i = 0; i < 16; ++i) {   // packed pair (key j, key j+16) -> adjacent slots
                h2 p; p[0] = (_Float16)w0[i]; p[1] = (_Float16)w1[i];
                *(h2*)&sVt[(cc + i) * VSTR + sv] = p;
            }
        }
        __syncthreads();

        // ---- S = Q * K^T : 4 key sub-tiles x 2 d-chunks ----
        v8f c0v, c1v, c2v, c3v;
        {
            v16h bk;
            #pragma unroll
            for (int h = 0; h < 16; ++h) bk[h] = sK[ m       * DD      + 16*hi + h];
            c0v = __builtin_amdgcn_wmma_f32_16x16x32_f16(false, aq0, false, bk, (short)0, v8f{}, false, false);
            #pragma unroll
            for (int h = 0; h < 16; ++h) bk[h] = sK[ m       * DD + 32 + 16*hi + h];
            c0v = __builtin_amdgcn_wmma_f32_16x16x32_f16(false, aq1, false, bk, (short)0, c0v, false, false);
            #pragma unroll
            for (int h = 0; h < 16; ++h) bk[h] = sK[(16 + m) * DD      + 16*hi + h];
            c1v = __builtin_amdgcn_wmma_f32_16x16x32_f16(false, aq0, false, bk, (short)0, v8f{}, false, false);
            #pragma unroll
            for (int h = 0; h < 16; ++h) bk[h] = sK[(16 + m) * DD + 32 + 16*hi + h];
            c1v = __builtin_amdgcn_wmma_f32_16x16x32_f16(false, aq1, false, bk, (short)0, c1v, false, false);
            #pragma unroll
            for (int h = 0; h < 16; ++h) bk[h] = sK[(32 + m) * DD      + 16*hi + h];
            c2v = __builtin_amdgcn_wmma_f32_16x16x32_f16(false, aq0, false, bk, (short)0, v8f{}, false, false);
            #pragma unroll
            for (int h = 0; h < 16; ++h) bk[h] = sK[(32 + m) * DD + 32 + 16*hi + h];
            c2v = __builtin_amdgcn_wmma_f32_16x16x32_f16(false, aq1, false, bk, (short)0, c2v, false, false);
            #pragma unroll
            for (int h = 0; h < 16; ++h) bk[h] = sK[(48 + m) * DD      + 16*hi + h];
            c3v = __builtin_amdgcn_wmma_f32_16x16x32_f16(false, aq0, false, bk, (short)0, v8f{}, false, false);
            #pragma unroll
            for (int h = 0; h < 16; ++h) bk[h] = sK[(48 + m) * DD + 32 + 16*hi + h];
            c3v = __builtin_amdgcn_wmma_f32_16x16x32_f16(false, aq1, false, bk, (short)0, c3v, false, false);
        }

        // ---- length mask: only the single partial tile pays (scalar-uniform branch) ----
        if (kb + KT > vlen) {
            if (kb + m >= vlen) {
                #pragma unroll
                for (int r = 0; r < 8; ++r) c0v[r] = -1.0e6f;
            }
            if (kb + 16 + m >= vlen) {
                #pragma unroll
                for (int r = 0; r < 8; ++r) c1v[r] = -1.0e6f;
            }
            if (kb + 32 + m >= vlen) {
                #pragma unroll
                for (int r = 0; r < 8; ++r) c2v[r] = -1.0e6f;
            }
            if (kb + 48 + m >= vlen) {
                #pragma unroll
                for (int r = 0; r < 8; ++r) c3v[r] = -1.0e6f;
            }
        }

        // ---- online softmax (exp2 domain); denominator comes from o4 = P*1 ----
        #pragma unroll
        for (int r = 0; r < 8; ++r) {
            float v = fmaxf(fmaxf(c0v[r], c1v[r]), fmaxf(c2v[r], c3v[r]));
            v = red_max16(v);
            const float mn    = fmaxf(m_run[r], v);
            const float alpha = __builtin_amdgcn_exp2f(m_run[r] - mn);
            m_run[r] = mn;
            c0v[r] = __builtin_amdgcn_exp2f(c0v[r] - mn);
            c1v[r] = __builtin_amdgcn_exp2f(c1v[r] - mn);
            c2v[r] = __builtin_amdgcn_exp2f(c2v[r] - mn);
            c3v[r] = __builtin_amdgcn_exp2f(c3v[r] - mn);
            o0[r] *= alpha; o1[r] *= alpha; o2[r] *= alpha; o3[r] *= alpha; o4[r] *= alpha;
        }

        // ---- P: C layout -> A layout via LDS, packed b32 stores in sigma slot order ----
        _Float16* pw = &sP[wave][0];
        #pragma unroll
        for (int r = 0; r < 8; ++r) {
            const int row = (r + 8 * hi) * KT;
            h2 p; p[0] = (_Float16)c0v[r]; p[1] = (_Float16)c1v[r];
            *(h2*)&pw[row + 2 * m] = p;               // slots 2m, 2m+1  (keys m, m+16)
            p[0] = (_Float16)c2v[r]; p[1] = (_Float16)c3v[r];
            *(h2*)&pw[row + 32 + 2 * m] = p;          // slots 32+2m,+1  (keys 32+m, 48+m)
        }
        asm volatile("s_wait_dscnt 0x0" ::: "memory");  // cross-lane RAW inside the wave

        // ---- O += P * V : 2 key-slot chunks x (4 d-tiles + ones tile) ----
        #pragma unroll
        for (int gg = 0; gg < 2; ++gg) {
            v16h ap;
            #pragma unroll
            for (int h = 0; h < 16; ++h) {
                const int k = 32 * gg + 16 * hi + h;
                ap[h] = pw[m * KT + k];
            }
            v16h bv;
            #pragma unroll
            for (int h = 0; h < 16; ++h) bv[h] = sVt[( 0 + m) * VSTR + 32*gg + 16*hi + h];
            o0 = __builtin_amdgcn_wmma_f32_16x16x32_f16(false, ap, false, bv, (short)0, o0, false, false);
            #pragma unroll
            for (int h = 0; h < 16; ++h) bv[h] = sVt[(16 + m) * VSTR + 32*gg + 16*hi + h];
            o1 = __builtin_amdgcn_wmma_f32_16x16x32_f16(false, ap, false, bv, (short)0, o1, false, false);
            #pragma unroll
            for (int h = 0; h < 16; ++h) bv[h] = sVt[(32 + m) * VSTR + 32*gg + 16*hi + h];
            o2 = __builtin_amdgcn_wmma_f32_16x16x32_f16(false, ap, false, bv, (short)0, o2, false, false);
            #pragma unroll
            for (int h = 0; h < 16; ++h) bv[h] = sVt[(48 + m) * VSTR + 32*gg + 16*hi + h];
            o3 = __builtin_amdgcn_wmma_f32_16x16x32_f16(false, ap, false, bv, (short)0, o3, false, false);
            // softmax denominator on the matrix pipe: row-sum of this P chunk
            o4 = __builtin_amdgcn_wmma_f32_16x16x32_f16(false, ap, false, ones, (short)0, o4, false, false);
        }
    }

    // ---- normalize + store fp32 (denominator = o4, any column) ----
    #pragma unroll
    for (int r = 0; r < 8; ++r) {
        const float inv = 1.0f / o4[r];
        float* op = Out + ((size_t)b * LL + qbase + r + 8 * hi) * DD + m;
        op[0]  = o0[r] * inv;
        op[16] = o1[r] * inv;
        op[32] = o2[r] * inv;
        op[48] = o3[r] * inv;
    }
}

extern "C" void kernel_launch(void* const* d_in, const int* in_sizes, int n_in,
                              void* d_out, int out_size, void* d_ws, size_t ws_size,
                              hipStream_t stream) {
    const float* Q  = (const float*)d_in[0];
    const float* K  = (const float*)d_in[1];
    const float* V  = (const float*)d_in[2];
    const int*   vl = (const int*)d_in[3];
    float* out = (float*)d_out;
    (void)in_sizes; (void)n_in; (void)out_size; (void)d_ws; (void)ws_size;
    dim3 grid(BB * (LL / 64));   // 1024 workgroups: one 64-row q-block each
    dim3 block(128);             // 4 waves (wave32)
    fa_fwd_kernel<<<grid, block, 0, stream>>>(Q, K, V, vl, out);
}
